// SimpleMoE_23081154249521
// MI455X (gfx1250) — matrix operand order repaired and checked by reading, not compiled, not run on hardware
//
#include <hip/hip_runtime.h>
#include <hip/hip_bf16.h>

// ---------------------------------------------------------------------------
// SimpleMoE for MI455X (gfx1250). Memory-bound dense MoE:
//   gates = softmax(x @ gate_w^T + gate_b)                  [N,E]
//   per expert e (serialized on stream):
//     w1T = bf16(transpose(w1[e]))   [H,D]   (one-time, ws)
//     w2T = bf16(transpose(w2[e]))   [D,H]   (one-time, ws)
//     h   = relu(x_bf16 @ w1[e] + b1[e])  -> bf16 ws        [N,H]
//     out (+)= gates[:,e] * (h @ w2[e] + b2[e])             [N,D]
// GEMMs: v_wmma_f32_16x16x32_bf16, 512 thr = 16 wave32 / block,
// tile 256x128x32, double-buffered LDS fed by GLOBAL_LOAD_ASYNC_TO_LDS_B128.
// ---------------------------------------------------------------------------

typedef __attribute__((ext_vector_type(16))) __bf16 v16bf;
typedef __attribute__((ext_vector_type(8)))  float  v8f;
typedef __attribute__((ext_vector_type(4)))  __bf16 v4bf;
typedef __attribute__((__vector_size__(16))) int    v4i;   // builtin's pointee

constexpr int N_ = 8192, D_ = 1024, H_ = 4096, E_ = 8;

constexpr int BM = 256;    // rows per block tile
constexpr int BN = 128;    // cols per block tile
constexpr int BK = 32;     // K per step (one wmma K)
constexpr int LDT = 48;    // LDS row stride in bf16 (96B: 16B-aligned, bank-spread)
constexpr int THREADS = 512;  // 16 waves

union FragAB { v16bf v; uint4 q[2]; };
union FragC  { v8f  v; float f[8]; };

// -------- gfx1250 async global->LDS path (guarded; falls back to sync copy) -
#if __has_builtin(__builtin_amdgcn_global_load_async_to_lds_b128)
#define ATHENA_ASYNC 1
#else
#define ATHENA_ASYNC 0
#endif

__device__ __forceinline__ void stage16(const __bf16* g, __bf16* l) {
#if ATHENA_ASYNC
  __builtin_amdgcn_global_load_async_to_lds_b128((v4i*)g, (v4i*)l, 0, 0);
#else
  *(uint4*)l = *(const uint4*)g;
#endif
}

__device__ __forceinline__ void wait_async() {
#if ATHENA_ASYNC
#if __has_builtin(__builtin_amdgcn_s_wait_asynccnt)
  __builtin_amdgcn_s_wait_asynccnt(0);
#else
  asm volatile("s_wait_asynccnt 0x0" ::: "memory");
#endif
#endif
}

// ---------------------------------------------------------------- gates ----
__global__ __launch_bounds__(256)
void moe_gates_kernel(const float* __restrict__ x,
                      const float* __restrict__ gw,
                      const float* __restrict__ gb,
                      float* __restrict__ gates) {
  int idx = blockIdx.x * blockDim.x + threadIdx.x;   // idx = n*E + e
  int n = idx >> 3;
  int e = idx & 7;
  const float4* xr = (const float4*)(x + (size_t)n * D_);
  const float4* wr = (const float4*)(gw + (size_t)e * D_);
  float acc = 0.f;
#pragma unroll 4
  for (int i = 0; i < D_ / 4; ++i) {
    float4 a = xr[i], b = wr[i];
    acc += a.x * b.x + a.y * b.y + a.z * b.z + a.w * b.w;
  }
  float logit = acc + gb[e];
  float mx = logit;
  for (int off = 1; off < 8; off <<= 1)
    mx = fmaxf(mx, __shfl_xor(mx, off, 8));
  float ex = __expf(logit - mx);
  float sm = ex;
  for (int off = 1; off < 8; off <<= 1)
    sm += __shfl_xor(sm, off, 8);
  gates[idx] = ex / sm;
}

// ------------------------------------------------------------- x -> bf16 ---
__global__ __launch_bounds__(256)
void cvt_x_kernel(const float* __restrict__ x, __bf16* __restrict__ xb) {
  int i = blockIdx.x * blockDim.x + threadIdx.x;     // one float4 -> 4 bf16
  float4 v = ((const float4*)x)[i];
  v4bf r;
  r.x = (__bf16)v.x; r.y = (__bf16)v.y; r.z = (__bf16)v.z; r.w = (__bf16)v.w;
  ((v4bf*)xb)[i] = r;
}

// -------------------------------------------- transpose + convert to bf16 --
// W [K x Ncols] f32 row-major  ->  WT [Ncols x K] bf16 row-major
__global__ __launch_bounds__(256)
void transpose_cvt_kernel(const float* __restrict__ W,
                          __bf16* __restrict__ WT, int K, int Ncols) {
  __shared__ float tile[32][33];
  const int n0 = blockIdx.x * 32;
  const int k0 = blockIdx.y * 32;
  const int tx = threadIdx.x & 31;
  const int ty = threadIdx.x >> 5;  // 0..7
#pragma unroll
  for (int j = 0; j < 32; j += 8)
    tile[ty + j][tx] = W[(size_t)(k0 + ty + j) * Ncols + n0 + tx];
  __syncthreads();
#pragma unroll
  for (int j = 0; j < 32; j += 8)
    WT[(size_t)(n0 + ty + j) * K + k0 + tx] = (__bf16)tile[tx][ty + j];
}

// ------------------------------------------------------------------ GEMM ---
// C[M,NDIM] = A[M,KDIM](bf16) * Bt[NDIM,KDIM](bf16, K-major) + bias
// RELU_STORE:  Hout = relu(C) as bf16
// else:        Cout (+)= gates[m*E+expert] * C   (f32; init selects = vs +=)
template<int KDIM, int NDIM, bool RELU_STORE>
__global__ __launch_bounds__(THREADS)
void moe_gemm_kernel(const __bf16* __restrict__ A,
                     const __bf16* __restrict__ Bt,
                     const float*  __restrict__ bias,
                     __bf16* __restrict__ Hout,
                     const float* __restrict__ gates,
                     float* __restrict__ Cout,
                     int expert, int init) {
  __shared__ __bf16 As[2][BM][LDT];   // 2 x 256 x 48 bf16 = 48 KB
  __shared__ __bf16 Bs[2][BN][LDT];   // 2 x 128 x 48 bf16 = 24 KB

  const int bm   = blockIdx.y * BM;
  const int bn   = blockIdx.x * BN;
  const int tid  = threadIdx.x;
  const int wave = tid >> 5;
  const int lane = tid & 31;
  const int wm   = (wave & 3) * 64;     // 4 waves along M, 64 rows each
  const int wn   = (wave >> 2) * 32;    // 4 waves along N, 32 cols each
  const int hl   = lane >> 4;           // lane half (0/1)
  const int lrow = lane & 15;

  const v8f vzero = {0.f, 0.f, 0.f, 0.f, 0.f, 0.f, 0.f, 0.f};
  FragC acc[4][2];
#pragma unroll
  for (int i = 0; i < 4; ++i)
#pragma unroll
    for (int j = 0; j < 2; ++j) acc[i][j].v = vzero;

  // stage one K-tile (pure bf16 16B chunks; async-to-LDS when available)
  auto stageA = [&](int k0, int buf) {
#pragma unroll
    for (int t = tid; t < (BM * BK) / 8; t += THREADS) {   // 1024 chunks
      int r = t >> 2, c = (t & 3) * 8;
      stage16(A + (size_t)(bm + r) * KDIM + k0 + c, &As[buf][r][c]);
    }
  };
  auto stageB = [&](int k0, int buf) {
#pragma unroll
    for (int t = tid; t < (BN * BK) / 8; t += THREADS) {   // 512 chunks
      int r = t >> 2, c = (t & 3) * 8;
      stage16(Bt + (size_t)(bn + r) * KDIM + k0 + c, &Bs[buf][r][c]);
    }
  };

  stageA(0, 0);
  stageB(0, 0);

  const int kA = hl * 8;    // A 16x32: lane<16 -> K[0..7],[16..23]; else +8
  const int kB = hl * 8;    // B 32x16: lane<16 -> K[0..7],[16..23]; else +8

  for (int k0 = 0, it = 0; k0 < KDIM; k0 += BK, ++it) {
    const int cur = it & 1;
    wait_async();        // my async stores into As/Bs[cur] have landed
    __syncthreads();     // everyone's have landed; prev reads of [cur^1] done
    if (k0 + BK < KDIM) {        // prefetch next tile into the other buffer
      stageA(k0 + BK, cur ^ 1);
      stageB(k0 + BK, cur ^ 1);
    }

    FragAB af[4], bfr[2];
#pragma unroll
    for (int ti = 0; ti < 4; ++ti) {
      const __bf16* src = &As[cur][wm + ti * 16 + lrow][0];
      af[ti].q[0] = *(const uint4*)(src + kA);
      af[ti].q[1] = *(const uint4*)(src + kA + 16);
    }
#pragma unroll
    for (int tj = 0; tj < 2; ++tj) {
      const __bf16* src = &Bs[cur][wn + tj * 16 + lrow][0];
      bfr[tj].q[0] = *(const uint4*)(src + kB);
      bfr[tj].q[1] = *(const uint4*)(src + kB + 16);
    }
#pragma unroll
    for (int ti = 0; ti < 4; ++ti)
#pragma unroll
      for (int tj = 0; tj < 2; ++tj)
        acc[ti][tj].v = __builtin_amdgcn_wmma_f32_16x16x32_bf16(
            false, af[ti].v, false, bfr[tj].v, (short)0, acc[ti][tj].v,
            false, false);
  }

  // ---- epilogue: C element r of lane -> (m = base + r + hl*8, n = lrow) ----
#pragma unroll
  for (int ti = 0; ti < 4; ++ti) {
#pragma unroll
    for (int tj = 0; tj < 2; ++tj) {
      const int gn = bn + wn + tj * 16 + lrow;
      const float bb = bias[gn];
#pragma unroll
      for (int r = 0; r < 8; ++r) {
        const int gm = bm + wm + ti * 16 + hl * 8 + r;
        float v = acc[ti][tj].f[r] + bb;
        if (RELU_STORE) {
          v = fmaxf(v, 0.f);
          Hout[(size_t)gm * NDIM + gn] = (__bf16)v;
        } else {
          const float g = gates[(size_t)gm * E_ + expert];
          float* dst = &Cout[(size_t)gm * NDIM + gn];
          if (init) *dst = g * v;
          else      *dst += g * v;
        }
      }
    }
  }
}

// ---------------------------------------------------------------- launch ---
extern "C" void kernel_launch(void* const* d_in, const int* in_sizes, int n_in,
                              void* d_out, int out_size, void* d_ws,
                              size_t ws_size, hipStream_t stream) {
  const float* x  = (const float*)d_in[0];
  const float* gw = (const float*)d_in[1];
  const float* gb = (const float*)d_in[2];
  const float* w1 = (const float*)d_in[3];
  const float* b1 = (const float*)d_in[4];
  const float* w2 = (const float*)d_in[5];
  const float* b2 = (const float*)d_in[6];
  float* out = (float*)d_out;

  // ws: gates 256KB (512KB slot) | xb 16MB | hb 64MB | w1T 8MB | w2T 8MB  (96.5 MB)
  char* ws = (char*)d_ws;
  float*  gates = (float*)ws;
  __bf16* xb  = (__bf16*)(ws + 512ull * 1024);
  __bf16* hb  = (__bf16*)(ws + 512ull * 1024 + 16ull * 1024 * 1024);
  __bf16* w1T = (__bf16*)(ws + 512ull * 1024 + 80ull * 1024 * 1024);
  __bf16* w2T = (__bf16*)(ws + 512ull * 1024 + 88ull * 1024 * 1024);

  moe_gates_kernel<<<(N_ * E_) / 256, 256, 0, stream>>>(x, gw, gb, gates);
  cvt_x_kernel<<<(N_ * D_ / 4) / 256, 256, 0, stream>>>(x, xb);

  for (int e = 0; e < E_; ++e) {
    const float* w1e = w1 + (size_t)e * D_ * H_;
    const float* b1e = b1 + (size_t)e * H_;
    const float* w2e = w2 + (size_t)e * H_ * D_;
    const float* b2e = b2 + (size_t)e * D_;

    // one-time per-expert weight transpose+convert (coalesced both sides)
    dim3 tg1(H_ / 32, D_ / 32);   // w1e [D,H] -> w1T [H,D]
    transpose_cvt_kernel<<<tg1, 256, 0, stream>>>(w1e, w1T, D_, H_);
    dim3 tg2(D_ / 32, H_ / 32);   // w2e [H,D] -> w2T [D,H]
    transpose_cvt_kernel<<<tg2, 256, 0, stream>>>(w2e, w2T, H_, D_);

    dim3 g1(H_ / BN, N_ / BM);    // (32, 32)
    moe_gemm_kernel<D_, H_, true><<<g1, THREADS, 0, stream>>>(
        xb, w1T, b1e, hb, nullptr, nullptr, e, 0);

    dim3 g2(D_ / BN, N_ / BM);    // (8, 32)
    moe_gemm_kernel<H_, D_, false><<<g2, THREADS, 0, stream>>>(
        hb, w2T, b2e, nullptr, gates, out, e, (e == 0) ? 1 : 0);
  }
}
